// HardLSTM_12876311954120
// MI455X (gfx1250) — matrix operand order repaired and checked by reading, not compiled, not run on hardware
//
#include <hip/hip_runtime.h>

// ---------------------------------------------------------------- constants
#define T_STEPS 2000
#define BATCH   32
#define INF_    320          // input features
#define HID     320          // hidden
#define GATES   1280         // 4*HID
#define NTG     80           // GATES/16 column tiles
#define KC1     10           // INF_/32  k-chunks (phase 1)
#define KC2     10           // HID/32   k-chunks (phase 2)
#define NWG     4            // persistent workgroups in scan
#define NT2     20           // column tiles per scan WG (4 gates * 5 jtiles)
#define JT      5            // hidden-unit tiles per WG (80/16)
#define HFRAG   (2 * KC2 * 32 * 16)   // halfs per h A-fragment buffer (10240)

typedef __attribute__((ext_vector_type(16))) _Float16 v16h;
typedef __attribute__((ext_vector_type(8)))  _Float16 v8h;
typedef __attribute__((ext_vector_type(8)))  float    v8f;

__device__ __forceinline__ v8f wmma16x16x32(v16h a, v16h b, v8f c) {
    return __builtin_amdgcn_wmma_f32_16x16x32_f16(false, a, false, b,
                                                  (short)0, c, false, false);
}

__device__ __forceinline__ v16h cat8(v8h lo, v8h hi) {
    return __builtin_shufflevector(lo, hi, 0, 1, 2, 3, 4, 5, 6, 7,
                                           8, 9, 10, 11, 12, 13, 14, 15);
}

__device__ __forceinline__ float hsig(float x) {
    return fminf(fmaxf(0.2f * x + 0.5f, 0.0f), 1.0f);
}
__device__ __forceinline__ float htanh(float x) {
    return fminf(fmaxf(x, -1.0f), 1.0f);
}

// ================================================================ phase 0
// one-time f32 -> f16 conversion (x, W_ih, W_hh); vectorized 8 elems/thread
__global__ __launch_bounds__(256)
void cvt_f32_f16(const float* __restrict__ src, _Float16* __restrict__ dst,
                 int n) {
    int i = (blockIdx.x * 256 + threadIdx.x) * 8;
    if (i + 8 <= n) {
        const float* s = src + i;
        v8h o;
        #pragma unroll
        for (int r = 0; r < 8; ++r) o[r] = (_Float16)s[r];
        *(v8h*)(dst + i) = o;
    }
}

// ================================================================ phase 1
// gates_ws[t][ntg][mt][lane][8] (f16, WMMA C-fragment order)
//   = f16( x[t] @ W_ih^T + b_ih + b_hh )
// No LDS: A and B fragments are loaded directly from the f16 copies with
// aligned contiguous per-lane loads (B: one v16h; A: two v8h + shuffle).
__global__ __launch_bounds__(256)
void hardlstm_gates_gemm(const _Float16* __restrict__ xh,
                         const _Float16* __restrict__ wih,
                         const float* __restrict__ b_ih,
                         const float* __restrict__ b_hh,
                         _Float16* __restrict__ gates_ws) {
    const int t    = blockIdx.x;
    const int cb   = blockIdx.y * 128;       // column base
    const int lane = threadIdx.x & 31;
    const int wave = threadIdx.x >> 5;       // ntile within the 128-col block
    const int half = lane >> 4;
    const int m0   = lane & 15;

    const _Float16* xrow0 = xh + (size_t)(t * BATCH + m0) * INF_;  // mt = 0
    const _Float16* xrow1 = xrow0 + 16 * INF_;                      // mt = 1
    const int col = cb + wave * 16 + m0;
    const _Float16* brow = wih + (size_t)col * INF_ + 16 * half;

    v8f acc0 = {}; v8f acc1 = {};
    #pragma unroll
    for (int kc = 0; kc < KC1; ++kc) {
        const int k0 = kc * 32 + 16 * half;
        v8h a0lo = *(const v8h*)(xrow0 + k0);
        v8h a0hi = *(const v8h*)(xrow0 + k0 + 8);
        v8h a1lo = *(const v8h*)(xrow1 + k0);
        v8h a1hi = *(const v8h*)(xrow1 + k0 + 8);
        v16h b   = *(const v16h*)(brow + kc * 32);
        acc0 = wmma16x16x32(cat8(a0lo, a0hi), b, acc0);
        acc1 = wmma16x16x32(cat8(a1lo, a1hi), b, acc1);
    }

    const float bias = b_ih[col] + b_hh[col];
    const int   ntg  = (cb + wave * 16) >> 4;

    v8h o0, o1;
    #pragma unroll
    for (int r = 0; r < 8; ++r) {
        o0[r] = (_Float16)(acc0[r] + bias);
        o1[r] = (_Float16)(acc1[r] + bias);
    }
    _Float16* base = gates_ws + (((size_t)t * NTG + ntg) * 2) * 256 + lane * 8;
    *(v8h*)(base)       = o0;   // mt = 0
    *(v8h*)(base + 256) = o1;   // mt = 1
}

// ================================================================ init
__global__ void hardlstm_scan_init(_Float16* __restrict__ h_frag,
                                   unsigned int* __restrict__ barrier_ct) {
    int tid = threadIdx.x;
    if (tid == 0) *barrier_ct = 0u;
    for (int i = tid; i < 2 * HFRAG; i += 256)
        h_frag[i] = (_Float16)0.0f;
}

// ================================================================ phase 2
// Persistent scan: NWG=4 workgroups, each 320 threads (10 waves).
// WG k owns hidden units [k*80, (k+1)*80): B-fragments of its 320 W_hh rows
// stay resident in LDS (205 KB). c state lives in VGPRs (v8f per wave).
// h is double-buffered in global scratch in A-fragment order: GEMM loads A
// directly with coalesced 32B per-lane loads; producer scatters with
// constant-offset b16 stores. One LDS exchange + one device barrier / step.
__global__ __launch_bounds__(320)
void hardlstm_scan(const _Float16* __restrict__ whh,
                   const _Float16* __restrict__ gates_ws,
                   float* __restrict__ out,
                   _Float16* __restrict__ h_frag,
                   unsigned int* __restrict__ barrier_ct) {
    extern __shared__ char smem[];
    _Float16* Bs = (_Float16*)smem;                    // [NT2][KC2][32][16] = 204800 B
    float*    Gs = (float*)(Bs + NT2 * KC2 * 32 * 16); // [NT2][2][32][8]   =  40960 B

    const int wg   = blockIdx.x;
    const int tid  = threadIdx.x;
    const int lane = tid & 31;
    const int wave = tid >> 5;           // 0..9

    // ---- one-time: copy this WG's W_hh slice into B-fragment LDS (v16h copies)
    for (int s = tid; s < NT2 * KC2 * 32; s += 320) {
        int ln = s & 31;
        int q  = s >> 5;
        int kc = q % KC2;
        int nt = q / KC2;
        int gate = nt / JT, jt = nt % JT;
        int col   = gate * HID + wg * 80 + jt * 16 + (ln & 15);
        int kbase = kc * 32 + 16 * (ln >> 4);
        *(v16h*)(Bs + s * 16) = *(const v16h*)(whh + (size_t)col * HID + kbase);
    }

    // elementwise-tile ownership (c state): tile (mtW, jtW)
    const int mtW = wave / JT;           // 0..1
    const int jtW = wave % JT;           // 0..4
    v8f c = {};

    // GEMM-tile ownership: column tiles {2*wave, 2*wave+1}, both M tiles
    const int nt0 = 2 * wave, nt1 = 2 * wave + 1;
    const int ntgA = (nt0 / JT) * 20 + wg * JT + (nt0 % JT);
    const int ntgB = (nt1 / JT) * 20 + wg * JT + (nt1 % JT);

    // producer: A-fragment scatter base for this lane's h values
    const int jW    = wg * 80 + jtW * 16 + (lane & 15);
    const int klW   = jW & 31;
    const int halfK = klW >> 4;
    const int eW    = klW & 15;
    const int hbase = (((mtW * KC2 + (jW >> 5)) * 32)
                       + halfK * 16 + 8 * (lane >> 4)) * 16 + eW;

    __syncthreads();

    for (int t = 0; t < T_STEPS; ++t) {
        const _Float16* hsrc = h_frag + (size_t)(t & 1) * HFRAG;
        const size_t tb = (size_t)t * NTG;

        // prefetch next step's gates_x fragments into cache
        if (t + 1 < T_STEPS) {
            const size_t tb1 = tb + NTG;
            __builtin_prefetch(gates_ws + ((tb1 + ntgA) * 2 + 0) * 256 + lane * 8, 0, 1);
            __builtin_prefetch(gates_ws + ((tb1 + ntgB) * 2 + 0) * 256 + lane * 8, 0, 1);
            __builtin_prefetch(gates_ws + ((tb1 + ntgA) * 2 + 1) * 256 + lane * 8, 0, 1);
            __builtin_prefetch(gates_ws + ((tb1 + ntgB) * 2 + 1) * 256 + lane * 8, 0, 1);
        }

        // ---- h @ W_hh^T : A straight from global (L2-hot, coalesced 32B/lane)
        v8f a00 = {}, a01 = {}, a10 = {}, a11 = {};
        #pragma unroll
        for (int kc = 0; kc < KC2; ++kc) {
            v16h A0 = *(const v16h*)(hsrc + ((0 * KC2 + kc) * 32 + lane) * 16);
            v16h A1 = *(const v16h*)(hsrc + ((1 * KC2 + kc) * 32 + lane) * 16);
            v16h B0 = *(const v16h*)(Bs + ((nt0 * KC2 + kc) * 32 + lane) * 16);
            v16h B1 = *(const v16h*)(Bs + ((nt1 * KC2 + kc) * 32 + lane) * 16);
            a00 = wmma16x16x32(A0, B0, a00);
            a01 = wmma16x16x32(A0, B1, a01);
            a10 = wmma16x16x32(A1, B0, a10);
            a11 = wmma16x16x32(A1, B1, a11);
        }

        // ---- add precomputed gates_x fragment, park gates in LDS
        {
            v8f* accs[4] = { &a00, &a01, &a10, &a11 };
            const int ntg[4] = { ntgA, ntgB, ntgA, ntgB };
            const int ntl[4] = { nt0,  nt1,  nt0,  nt1  };
            const int mts[4] = { 0, 0, 1, 1 };
            #pragma unroll
            for (int u = 0; u < 4; ++u) {
                v8h gx = *(const v8h*)(gates_ws +
                          ((tb + ntg[u]) * 2 + mts[u]) * 256 + lane * 8);
                v8f s;
                #pragma unroll
                for (int r = 0; r < 8; ++r) s[r] = (*accs[u])[r] + (float)gx[r];
                *(v8f*)(Gs + ((ntl[u] * 2 + mts[u]) * 32 + lane) * 8) = s;
            }
        }
        __syncthreads();

        // ---- elementwise state update (c resident in VGPRs)
        {
            const int fb = (jtW * 2 + mtW) * 32 + lane;   // per-gate frag slot
            v8f vi = *(const v8f*)(Gs + ((0 * JT * 2) * 32) * 8 + fb * 8);
            v8f vf = *(const v8f*)(Gs + ((1 * JT * 2) * 32) * 8 + fb * 8);
            v8f vg = *(const v8f*)(Gs + ((2 * JT * 2) * 32) * 8 + fb * 8);
            v8f vo = *(const v8f*)(Gs + ((3 * JT * 2) * 32) * 8 + fb * 8);

            _Float16* hdst = h_frag + (size_t)((t + 1) & 1) * HFRAG;
            float* orow = out + (size_t)t * BATCH * HID + jW;

            #pragma unroll
            for (int r = 0; r < 8; ++r) {
                float ig = hsig(vi[r]);
                float fg = hsig(vf[r]);
                float gg = htanh(vg[r]);
                float og = hsig(vo[r]);
                c[r] = fg * c[r] + ig * gg;
                float h = og * htanh(c[r]);
                int b = mtW * 16 + 8 * (lane >> 4) + r;
                orow[(size_t)b * HID] = h;
                hdst[hbase + r * 16] = (_Float16)h;   // A-frag scatter, const offsets
            }
        }
        __threadfence();
        __syncthreads();

        // ---- cross-WGP step barrier (device-scope atomic spin)
        if (tid == 0) {
            __hip_atomic_fetch_add(barrier_ct, 1u, __ATOMIC_ACQ_REL,
                                   __HIP_MEMORY_SCOPE_AGENT);
            const unsigned target = (unsigned)(t + 1) * NWG;
            while (__hip_atomic_load(barrier_ct, __ATOMIC_ACQUIRE,
                                     __HIP_MEMORY_SCOPE_AGENT) < target) {
                __builtin_amdgcn_s_sleep(1);
            }
        }
        __syncthreads();
        __threadfence();
    }
}

// ================================================================ launch
extern "C" void kernel_launch(void* const* d_in, const int* in_sizes, int n_in,
                              void* d_out, int out_size, void* d_ws, size_t ws_size,
                              hipStream_t stream) {
    const float* x    = (const float*)d_in[0];
    const float* w_ih = (const float*)d_in[1];
    const float* w_hh = (const float*)d_in[2];
    const float* b_ih = (const float*)d_in[3];
    const float* b_hh = (const float*)d_in[4];
    float* out = (float*)d_out;

    // ws layout: [gates_ws | h_frag x2 | x_f16 | wih_f16 | whh_f16 | counter]
    const size_t n_x   = (size_t)T_STEPS * BATCH * INF_;   // 20.48M
    const size_t n_w   = (size_t)GATES * INF_;             // 409600
    _Float16* gates_ws = (_Float16*)d_ws;
    const size_t gates_elems = (size_t)T_STEPS * NTG * 2 * 32 * 8;  // 81.92M halfs
    _Float16* h_frag  = gates_ws + gates_elems;
    _Float16* x_f16   = h_frag + 2 * HFRAG;
    _Float16* wih_f16 = x_f16 + n_x;
    _Float16* whh_f16 = wih_f16 + n_w;
    uintptr_t p = (uintptr_t)(whh_f16 + n_w);
    p = (p + 127) & ~(uintptr_t)127;
    unsigned int* barrier_ct = (unsigned int*)p;

    // phase 0: f16 copies of x / W_ih / W_hh
    cvt_f32_f16<<<(int)(n_x / 2048), 256, 0, stream>>>(x, x_f16, (int)n_x);
    cvt_f32_f16<<<(int)(n_w / 2048), 256, 0, stream>>>(w_ih, wih_f16, (int)n_w);
    cvt_f32_f16<<<(int)(n_w / 2048), 256, 0, stream>>>(w_hh, whh_f16, (int)n_w);

    // phase 1: input projection GEMM (fills gates_ws in fragment order)
    dim3 g1(T_STEPS, GATES / 128);
    hardlstm_gates_gemm<<<g1, 256, 0, stream>>>(x_f16, wih_f16, b_ih, b_hh,
                                                gates_ws);

    // init h / barrier state (fresh every launch: graph-replay safe)
    hardlstm_scan_init<<<1, 256, 0, stream>>>(h_frag, barrier_ct);

    // phase 2: persistent recurrent scan, W_hh LDS-resident
    const size_t smem2 = (size_t)(NT2 * KC2 * 32 * 16) * sizeof(_Float16)
                       + (size_t)NT2 * 2 * 32 * 8 * sizeof(float);  // 245760 B
    hardlstm_scan<<<NWG, 320, smem2, stream>>>(whh_f16, gates_ws, out,
                                               h_frag, barrier_ct);
}
